// QuantizedLinear_80985903334133
// MI455X (gfx1250) — matrix operand order repaired and checked by reading, not compiled, not run on hardware
//
#include <hip/hip_runtime.h>
#include <hip/hip_bf16.h>

#define IN_FEATURES 8192
#define OUT_FEATURES 8192
#define CODESZ 8
#define TOKENS 32
#define KSPLIT 8

typedef __attribute__((ext_vector_type(16))) __bf16 v16bf;
typedef __attribute__((ext_vector_type(8)))  float  v8f;

union Frag16 {
    v16bf v;
    unsigned int u[8];
};

__device__ __forceinline__ unsigned int pk_bf16(float a, float b) {
    union { __bf16 h[2]; unsigned int u; } t;
    t.h[0] = (__bf16)a;
    t.h[1] = (__bf16)b;
    return t.u;
}

// ---------------------------------------------------------------------------
// Kernel 1: xh = FWHT(x * SU) / sqrt(n), emitted as bf16 (512 KB, stays in L2)
// One workgroup per token; 8192 floats of LDS; 13 butterfly stages.
// ---------------------------------------------------------------------------
__global__ void __launch_bounds__(256)
fwht_in_kernel(const float* __restrict__ x,
               const float* __restrict__ SU,
               unsigned short* __restrict__ xh) {
    __shared__ float s[IN_FEATURES];
    const int t   = blockIdx.x;
    const int tid = threadIdx.x;

    const float* xr = x + (size_t)t * IN_FEATURES;
    for (int i = tid; i < IN_FEATURES; i += 256)
        s[i] = xr[i] * SU[i];
    __syncthreads();

    for (int h = 1; h < IN_FEATURES; h <<= 1) {
        for (int p = tid; p < IN_FEATURES / 2; p += 256) {
            const int i = ((p & ~(h - 1)) << 1) | (p & (h - 1));
            const int j = i + h;
            const float a = s[i], b = s[j];
            s[i] = a + b;
            s[j] = a - b;
        }
        __syncthreads();
    }

    const float inv = 0.011048543456039805f;  // 1/sqrt(8192)
    unsigned short* orow = xh + (size_t)t * IN_FEATURES;
    for (int i = tid; i < IN_FEATURES; i += 256) {
        union { __bf16 h; unsigned short u; } c;
        c.h = (__bf16)(s[i] * inv);
        orow[i] = c.u;
    }
}

// ---------------------------------------------------------------------------
// Kernel 2: on-the-fly codebook decode + bf16 WMMA GEMM.
//   Each wave owns one N-tile of 16 output features and one K-slice of 1024.
//   Per K-step of 32: each lane's WMMA B fragment (col N=lane%16,
//   K = k0 + 16*(lane/16) .. +15) is exactly TWO codebook entries -> decode
//   lands directly in WMMA register layout, no shuffles, no LDS staging.
//   A fragments come from bf16 xh in L2 (two 16B loads per M-tile).
// ---------------------------------------------------------------------------
__global__ void __launch_bounds__(256)
decode_gemm_kernel(const unsigned short* __restrict__ xh,
                   const float* __restrict__ codebook,
                   const int* __restrict__ Qidxs,
                   float* __restrict__ zpart) {
    const int lane   = threadIdx.x & 31;
    const int wave   = threadIdx.x >> 5;
    const int ntile  = blockIdx.x * 8 + wave;        // 0..511
    const int kslice = blockIdx.y;                   // 0..KSPLIT-1
    const int hi     = lane >> 4;                    // half-wave select
    const int n      = ntile * 16 + (lane & 15);     // output feature

    const int kBeg = kslice * (IN_FEATURES / KSPLIT);
    const int kEnd = kBeg + (IN_FEATURES / KSPLIT);

    v8f acc0 = {};  // tokens 0..15
    v8f acc1 = {};  // tokens 16..31

    const int m0 = lane & 15;
    const unsigned short* arow0 = xh + (size_t)m0 * IN_FEATURES;
    const unsigned short* arow1 = xh + (size_t)(m0 + 16) * IN_FEATURES;
    const int*    qrow = Qidxs + (size_t)n * (IN_FEATURES / CODESZ);
    const float4* cb   = reinterpret_cast<const float4*>(codebook);

    for (int k0 = kBeg; k0 < kEnd; k0 += 32) {
        // ---- A fragments (bf16 xh, contiguous K = k0 + 16*(lane/16) + s, same order as B) ----
        const int ka = k0 + (hi << 4);  // lanes<16: k0, lanes>=16: k0+16
        const uint4 lo0 = *reinterpret_cast<const uint4*>(arow0 + ka);
        const uint4 hi0 = *reinterpret_cast<const uint4*>(arow0 + ka + 8);
        const uint4 lo1 = *reinterpret_cast<const uint4*>(arow1 + ka);
        const uint4 hi1 = *reinterpret_cast<const uint4*>(arow1 + ka + 8);
        Frag16 a0, a1;
        a0.u[0] = lo0.x; a0.u[1] = lo0.y; a0.u[2] = lo0.z; a0.u[3] = lo0.w;
        a0.u[4] = hi0.x; a0.u[5] = hi0.y; a0.u[6] = hi0.z; a0.u[7] = hi0.w;
        a1.u[0] = lo1.x; a1.u[1] = lo1.y; a1.u[2] = lo1.z; a1.u[3] = lo1.w;
        a1.u[4] = hi1.x; a1.u[5] = hi1.y; a1.u[6] = hi1.z; a1.u[7] = hi1.w;

        // ---- B fragment: decode 2 codebook entries (16 contiguous K) ----
        const int q    = (k0 >> 3) + (hi << 1);
        const int idx0 = qrow[q];
        const int idx1 = qrow[q + 1];
        __builtin_prefetch(qrow + q + 32, 0, 1);  // next index cachelines

        const float4 c0 = cb[2 * idx0];
        const float4 c1 = cb[2 * idx0 + 1];
        const float4 c2 = cb[2 * idx1];
        const float4 c3 = cb[2 * idx1 + 1];
        Frag16 b;
        b.u[0] = pk_bf16(c0.x, c0.y); b.u[1] = pk_bf16(c0.z, c0.w);
        b.u[2] = pk_bf16(c1.x, c1.y); b.u[3] = pk_bf16(c1.z, c1.w);
        b.u[4] = pk_bf16(c2.x, c2.y); b.u[5] = pk_bf16(c2.z, c2.w);
        b.u[6] = pk_bf16(c3.x, c3.y); b.u[7] = pk_bf16(c3.z, c3.w);

        // D = A x B + C   (16x16x32 bf16 -> f32)
        acc0 = __builtin_amdgcn_wmma_f32_16x16x32_bf16(
            false, a0.v, false, b.v, (short)0, acc0, false, false);
        acc1 = __builtin_amdgcn_wmma_f32_16x16x32_bf16(
            false, a1.v, false, b.v, (short)0, acc1, false, false);
    }

    // ---- store D tiles (ISA 32-bit C/D layout) into this K-slice partial ----
    float* zp = zpart + (size_t)kslice * TOKENS * OUT_FEATURES;
    const int mbase = hi << 3;  // lanes<16: M=v, lanes>=16: M=v+8
#pragma unroll
    for (int v = 0; v < 8; ++v) {
        zp[(size_t)(mbase + v) * OUT_FEATURES + n]      = acc0[v];
        zp[(size_t)(16 + mbase + v) * OUT_FEATURES + n] = acc1[v];
    }
}

// ---------------------------------------------------------------------------
// Kernel 3: sum K-slice partials, FWHT over outputs, scale by SV*Wscale/sqrt(n)
// ---------------------------------------------------------------------------
__global__ void __launch_bounds__(256)
fwht_out_kernel(const float* __restrict__ zpart,
                const float* __restrict__ SV,
                const float* __restrict__ Wscale,
                float* __restrict__ out) {
    __shared__ float s[OUT_FEATURES];
    const int t   = blockIdx.x;
    const int tid = threadIdx.x;

    for (int i = tid; i < OUT_FEATURES; i += 256) {
        float acc = 0.f;
#pragma unroll
        for (int k = 0; k < KSPLIT; ++k)
            acc += zpart[(size_t)k * TOKENS * OUT_FEATURES +
                         (size_t)t * OUT_FEATURES + i];
        s[i] = acc;
    }
    __syncthreads();

    for (int h = 1; h < OUT_FEATURES; h <<= 1) {
        for (int p = tid; p < OUT_FEATURES / 2; p += 256) {
            const int i = ((p & ~(h - 1)) << 1) | (p & (h - 1));
            const int j = i + h;
            const float a = s[i], b = s[j];
            s[i] = a + b;
            s[j] = a - b;
        }
        __syncthreads();
    }

    const float inv = 0.011048543456039805f;  // 1/sqrt(8192)
    const float ws  = Wscale[0];
    float* orow = out + (size_t)t * OUT_FEATURES;
    for (int i = tid; i < OUT_FEATURES; i += 256)
        orow[i] = s[i] * inv * SV[i] * ws;
}

// ---------------------------------------------------------------------------
extern "C" void kernel_launch(void* const* d_in, const int* in_sizes, int n_in,
                              void* d_out, int out_size, void* d_ws, size_t ws_size,
                              hipStream_t stream) {
    const float* x        = (const float*)d_in[0];
    const float* codebook = (const float*)d_in[1];
    const int*   Qidxs    = (const int*)  d_in[2];
    const float* SU       = (const float*)d_in[3];
    const float* SV       = (const float*)d_in[4];
    const float* Wscale   = (const float*)d_in[5];
    float*       out      = (float*)d_out;

    // workspace: [bf16 xh: 512 KB][fp32 z partials: KSPLIT * 1 MB]
    unsigned short* xh = (unsigned short*)d_ws;
    float* zpart = (float*)((char*)d_ws +
                            (size_t)TOKENS * IN_FEATURES * sizeof(unsigned short));

    fwht_in_kernel<<<TOKENS, 256, 0, stream>>>(x, SU, xh);

    dim3 g2(OUT_FEATURES / 16 / 8, KSPLIT);  // 64 x 8 blocks, 8 waves each
    decode_gemm_kernel<<<g2, 256, 0, stream>>>(xh, codebook, Qidxs, zpart);

    fwht_out_kernel<<<TOKENS, 256, 0, stream>>>(zpart, SV, Wscale, out);
}
